// MLA_83966610637436
// MI455X (gfx1250) — matrix operand order repaired and checked by reading, not compiled, not run on hardware
//
#include <hip/hip_runtime.h>
#include <hip/hip_bf16.h>

#define BB 2
#define SS 2048
#define EE 128
#define HH 8
#define NROWS (BB * SS)   // 4096
#define NCOLS (EE * HH)   // 1024
#define SCL 0.08838834764831845f  // 128^-0.5

typedef __attribute__((ext_vector_type(16))) __bf16 v16bf;
typedef __attribute__((ext_vector_type(8)))  float  v8f;

union Frag { v16bf v; unsigned d[8]; };

// float -> bf16 (round to nearest even), bit-level
__device__ __forceinline__ unsigned short f2bf(float f) {
  unsigned u = __float_as_uint(f);
  u += 0x7FFFu + ((u >> 16) & 1u);
  return (unsigned short)(u >> 16);
}
__device__ __forceinline__ unsigned pk2(float a, float b) {
  return (unsigned)f2bf(a) | ((unsigned)f2bf(b) << 16);
}
// dword load at (uniform base + 32-bit byte offset [+ constant imm])
__device__ __forceinline__ unsigned ldb(const void* base, unsigned boff) {
  return *(const unsigned*)((const char*)base + boff);
}
// Constant byte offsets within a fragment (lane-half part folded into base off):
// A fragment (16x32 bf16): dword v -> elems 2*v                      [+16 if hi half]
// B fragment (32x16 bf16): dword v -> elems 2*v                      [+16 if hi half]
#define AOFF(v) (4u * (v))
#define BOFF(v) (4u * (v))

__device__ __forceinline__ v8f wmma_bf16(const Frag& a, const Frag& b, v8f c) {
  return __builtin_amdgcn_wmma_f32_16x16x32_bf16(false, a.v, false, b.v,
                                                 (short)0, c, false, false);
}

// ---------------------------------------------------------------------------
// Kernel 0: bulk fp32 -> packed bf16 conversion (2 elements / thread).
// ---------------------------------------------------------------------------
__global__ void mla_cvt(const float* __restrict__ src, unsigned* __restrict__ dst,
                        int n2) {
  int i = blockIdx.x * blockDim.x + threadIdx.x;
  if (i < n2) {
    float2 f = ((const float2*)src)[i];
    dst[i] = pk2(f.x, f.y);
  }
}

// ---------------------------------------------------------------------------
// Kernel 1: K/Q/V projections (bf16 in, bf16 out).  One wave = 16 rows x 64 cols.
// A fragment shared across K/Q/V: 12 WMMA per k-step, 4 k-steps.
// ---------------------------------------------------------------------------
__global__ __launch_bounds__(32, 1)
void mla_qkv(const unsigned short* __restrict__ xbf,
             const unsigned short* __restrict__ Wkb, const float* __restrict__ bk,
             const unsigned short* __restrict__ Wqb, const float* __restrict__ bq,
             const unsigned short* __restrict__ Wvb, const float* __restrict__ bv,
             unsigned short* __restrict__ Kbf,
             unsigned short* __restrict__ Qbf,
             unsigned short* __restrict__ Vt) {
  const int lane = threadIdx.x;
  const int n = lane & 15, half = lane >> 4;
  const int mT = blockIdx.x * 16;
  const int nT = blockIdx.y * 64;
  const unsigned ha = half ? 32u : 0u;   // A-frag hi-half byte offset
  const unsigned hb = half ? 32u : 0u;   // B-frag hi-half byte offset
  const unsigned xo = (unsigned)(mT + n) * (EE * 2) + ha;  // per-lane A base (bytes)
  const unsigned wo = (unsigned)(nT + n) * (EE * 2) + hb;  // per-lane B base (bytes)
  v8f cK[4], cQ[4], cV[4];
#pragma unroll
  for (int s = 0; s < 4; ++s) { cK[s] = {}; cQ[s] = {}; cV[s] = {}; }
#pragma unroll
  for (int kb = 0; kb < EE * 2; kb += 64) {  // k-step = 32 elems = 64 bytes
    Frag a;
#pragma unroll
    for (int v = 0; v < 8; ++v)
      a.d[v] = ldb(xbf, xo + (unsigned)kb + AOFF(v));
#pragma unroll
    for (int s = 0; s < 4; ++s) {
      Frag fk, fq, fv;
#pragma unroll
      for (int v = 0; v < 8; ++v) {
        const unsigned c = (unsigned)kb + (unsigned)s * 4096u + BOFF(v);  // s*16 rows
        fk.d[v] = ldb(Wkb, wo + c);
        fq.d[v] = ldb(Wqb, wo + c);
        fv.d[v] = ldb(Wvb, wo + c);
      }
      cK[s] = wmma_bf16(a, fk, cK[s]);
      cQ[s] = wmma_bf16(a, fq, cQ[s]);
      cV[s] = wmma_bf16(a, fv, cV[s]);
    }
  }
#pragma unroll
  for (int s = 0; s < 4; ++s) {
    int nG = nT + s * 16 + n;
    int h = nG & (HH - 1), e = nG >> 3;  // reshape(E,H): head is fastest dim
    float bK = bk[nG], bQ = bq[nG], bV = bv[nG];
#pragma unroll
    for (int v = 0; v < 8; ++v) {
      int row = mT + v + 8 * half;       // C-layout: M = v + 8*half
      int b = row >> 11;
      int si = row & (SS - 1);
      size_t bh = (size_t)(b * HH + h);
      Kbf[(bh * SS + si) * EE + e] = f2bf(cK[s][v] + bK);
      Qbf[(bh * SS + si) * EE + e] = f2bf(cQ[s][v] + bQ);
      Vt[(bh * EE + e) * SS + si]  = f2bf(cV[s][v] + bV);
    }
  }
}

// ---------------------------------------------------------------------------
// Kernel 2: column-wise softmax stats (axis=2 => normalize over i).
// One wave per (b*h, 16-col j tile); 2 independent i-tiles per iteration so
// the two WMMA accumulation chains hide each other's RAW-hazard NOPs.
// ---------------------------------------------------------------------------
__global__ __launch_bounds__(32, 1)
void mla_colstats(const unsigned short* __restrict__ Kbf,
                  const unsigned short* __restrict__ Qbf,
                  float* __restrict__ mS, float* __restrict__ dS) {
  const int lane = threadIdx.x;
  const int n = lane & 15, half = lane >> 4;
  const int bh = blockIdx.x;
  const int jT = blockIdx.y * 16;
  const unsigned short* Kb = Kbf + (size_t)bh * SS * EE;
  const unsigned short* Qb = Qbf + (size_t)bh * SS * EE;
  const unsigned ha = half ? 32u : 0u;
  const unsigned hb = half ? 32u : 0u;
  const unsigned qo = (unsigned)(jT + n) * (EE * 2) + hb;
  Frag bq[4];  // Q^T tile, invariant over the i loop
#pragma unroll
  for (int kb = 0; kb < 4; ++kb)
#pragma unroll
    for (int v = 0; v < 8; ++v)
      bq[kb].d[v] = ldb(Qb, qo + (unsigned)kb * 64u + BOFF(v));
  float mloc = -1e30f, dloc = 0.f;
  unsigned ko = (unsigned)n * (EE * 2) + ha;  // running K row offset (bytes)
  for (int iT = 0; iT < SS; iT += 32, ko += 32u * EE * 2) {
    __builtin_prefetch((const char*)Kb + ko + 32u * EE * 2, 0, 1);
    v8f c0 = {}, c1 = {};
#pragma unroll
    for (int kb = 0; kb < 4; ++kb) {
      Frag a0, a1;
#pragma unroll
      for (int v = 0; v < 8; ++v) {
        a0.d[v] = ldb(Kb, ko + (unsigned)kb * 64u + AOFF(v));
        a1.d[v] = ldb(Kb, ko + 4096u + (unsigned)kb * 64u + AOFF(v));  // +16 rows
      }
      c0 = wmma_bf16(a0, bq[kb], c0);
      c1 = wmma_bf16(a1, bq[kb], c1);
    }
#pragma unroll
    for (int v = 0; v < 8; ++v) {  // lane holds fixed column, 8+8 rows
      float s0 = c0[v] * SCL, s1 = c1[v] * SCL;
      float mn = fmaxf(mloc, fmaxf(s0, s1));
      dloc = dloc * __expf(mloc - mn) + __expf(s0 - mn) + __expf(s1 - mn);
      mloc = mn;
    }
  }
  float mo = __shfl_xor(mloc, 16, 32);
  float dq = __shfl_xor(dloc, 16, 32);
  float mn = fmaxf(mloc, mo);
  float dn = dloc * __expf(mloc - mn) + dq * __expf(mo - mn);
  if (lane < 16) {
    mS[(size_t)bh * SS + jT + n] = mn;
    dS[(size_t)bh * SS + jT + n] = dn;
  }
}

// ---------------------------------------------------------------------------
// Kernel 3: o[i,e] = sum_j exp(s[i,j]*scl - m[j]) / D[j] * v[j,e].
// One wave per (b*h, 16-row i tile); 32 j per iter; P transposed via LDS.
// ---------------------------------------------------------------------------
__global__ __launch_bounds__(32, 1)
void mla_attnav(const unsigned short* __restrict__ Kbf,
                const unsigned short* __restrict__ Qbf,
                const unsigned short* __restrict__ Vt,
                const float* __restrict__ mS,
                const float* __restrict__ dS,
                unsigned short* __restrict__ Obf) {
  __shared__ unsigned short lp[16 * 32];
  const int lane = threadIdx.x;
  const int n = lane & 15, half = lane >> 4;
  const int bh = blockIdx.x;
  const int iT = blockIdx.y * 16;
  const unsigned short* Kb = Kbf + (size_t)bh * SS * EE;
  const unsigned short* Qb = Qbf + (size_t)bh * SS * EE;
  const unsigned short* Vb = Vt + (size_t)bh * EE * SS;
  const float* mRow = mS + (size_t)bh * SS;
  const float* dRow = dS + (size_t)bh * SS;
  const unsigned ha = half ? 32u : 0u;
  const unsigned hb = half ? 32u : 0u;
  Frag aK[4];  // K rows for this i tile, invariant over j loop
  const unsigned ko = (unsigned)(iT + n) * (EE * 2) + ha;
#pragma unroll
  for (int kb = 0; kb < 4; ++kb)
#pragma unroll
    for (int v = 0; v < 8; ++v)
      aK[kb].d[v] = ldb(Kb, ko + (unsigned)kb * 64u + AOFF(v));
  v8f o[8];
#pragma unroll
  for (int t = 0; t < 8; ++t) o[t] = {};
  unsigned qo = (unsigned)n * (EE * 2) + hb;  // running Q row offset (bytes)
  unsigned vo = (unsigned)n * (SS * 2) + hb;  // running V^T offset (bytes)
  for (int jB = 0; jB < SS; jB += 32, qo += 32u * EE * 2, vo += 64u) {
    v8f c0 = {}, c1 = {};
#pragma unroll
    for (int kb = 0; kb < 4; ++kb) {
      Frag q0, q1;
#pragma unroll
      for (int v = 0; v < 8; ++v) {
        q0.d[v] = ldb(Qb, qo + (unsigned)kb * 64u + BOFF(v));
        q1.d[v] = ldb(Qb, qo + 4096u + (unsigned)kb * 64u + BOFF(v));  // +16 rows
      }
      c0 = wmma_bf16(aK[kb], q0, c0);
      c1 = wmma_bf16(aK[kb], q1, c1);
    }
    float m0 = mRow[jB + n],      id0 = 1.0f / dRow[jB + n];
    float m1 = mRow[jB + 16 + n], id1 = 1.0f / dRow[jB + 16 + n];
#pragma unroll
    for (int v = 0; v < 8; ++v) {  // write P tile (16 x 32, row-major) to LDS
      int r = v + 8 * half;
      lp[r * 32 + n]      = f2bf(__expf(c0[v] * SCL - m0) * id0);
      lp[r * 32 + 16 + n] = f2bf(__expf(c1[v] * SCL - m1) * id1);
    }
    __syncthreads();
    Frag ap;  // reload P in A-fragment layout (lane row = n, ld = 32 elems)
#pragma unroll
    for (int v = 0; v < 8; ++v)
      ap.d[v] = *(const unsigned*)((const char*)lp + (unsigned)n * 64u + ha + AOFF(v));
    __syncthreads();
#pragma unroll
    for (int t = 0; t < 8; ++t) {  // B = V^T (B,H,E,S): e-tile t at imm t*65536
      Frag bv;
#pragma unroll
      for (int v = 0; v < 8; ++v)
        bv.d[v] = ldb(Vb, vo + (unsigned)t * 65536u + BOFF(v));
      o[t] = wmma_bf16(ap, bv, o[t]);
    }
  }
  int bb = bh >> 3, h = bh & 7;
#pragma unroll
  for (int t = 0; t < 8; ++t)
#pragma unroll
    for (int v = 0; v < 8; ++v) {
      int r = iT + v + 8 * half;
      Obf[(size_t)(bb * SS + r) * NCOLS + h * EE + t * 16 + n] = f2bf(o[t][v]);
    }
}

// ---------------------------------------------------------------------------
// Kernel 4: out = O_concat @ Wp^T + bp   (4096 x 128, K = 1024), bf16 weights.
// ---------------------------------------------------------------------------
__global__ __launch_bounds__(32, 1)
void mla_outproj(const unsigned short* __restrict__ Obf,
                 const unsigned short* __restrict__ Wpb,
                 const float* __restrict__ bp,
                 float* __restrict__ out) {
  const int lane = threadIdx.x;
  const int n = lane & 15, half = lane >> 4;
  const int mT = blockIdx.x * 16;
  const int nT = blockIdx.y * 64;
  const unsigned ha = half ? 32u : 0u;
  const unsigned hb = half ? 32u : 0u;
  const unsigned oo = (unsigned)(mT + n) * (NCOLS * 2) + ha;
  const unsigned wo = (unsigned)(nT + n) * (NCOLS * 2) + hb;
  v8f c[4];
#pragma unroll
  for (int s = 0; s < 4; ++s) c[s] = {};
  for (int kb = 0; kb < NCOLS * 2; kb += 64) {  // 32 elems = 64 bytes per step
    Frag a;
#pragma unroll
    for (int v = 0; v < 8; ++v)
      a.d[v] = ldb(Obf, oo + (unsigned)kb + AOFF(v));
#pragma unroll
    for (int s = 0; s < 4; ++s) {
      Frag bw;
#pragma unroll
      for (int v = 0; v < 8; ++v)
        bw.d[v] = ldb(Wpb, wo + (unsigned)kb + (unsigned)s * 32768u + BOFF(v));
      c[s] = wmma_bf16(a, bw, c[s]);
    }
  }
#pragma unroll
  for (int s = 0; s < 4; ++s) {
    int nG = nT + s * 16 + n;
    float bb = bp[nG];
#pragma unroll
    for (int v = 0; v < 8; ++v) {
      int row = mT + v + 8 * half;
      out[(size_t)row * EE + nG] = c[s][v] + bb;
    }
  }
}

// ---------------------------------------------------------------------------
extern "C" void kernel_launch(void* const* d_in, const int* in_sizes, int n_in,
                              void* d_out, int out_size, void* d_ws, size_t ws_size,
                              hipStream_t stream) {
  const float* x  = (const float*)d_in[0];
  const float* Wk = (const float*)d_in[1];
  const float* bk = (const float*)d_in[2];
  const float* Wq = (const float*)d_in[3];
  const float* bq = (const float*)d_in[4];
  const float* Wv = (const float*)d_in[5];
  const float* bv = (const float*)d_in[6];
  const float* Wp = (const float*)d_in[7];
  const float* bp = (const float*)d_in[8];
  float* out = (float*)d_out;

  char* ws = (char*)d_ws;
  const size_t MB = (size_t)1 << 20;
  unsigned short* Kbf = (unsigned short*)(ws + 0 * MB);    // (B,H,S,E) bf16, 8MB
  unsigned short* Qbf = (unsigned short*)(ws + 8 * MB);    // (B,H,S,E) bf16, 8MB
  unsigned short* Vt  = (unsigned short*)(ws + 16 * MB);   // (B,H,E,S) bf16, 8MB
  unsigned short* Obf = (unsigned short*)(ws + 24 * MB);   // (B,S,H*E) bf16, 8MB
  float* mS = (float*)(ws + 32 * MB);                      // (B*H,S) f32
  float* dS = mS + (size_t)BB * HH * SS;
  unsigned short* xbf = (unsigned short*)(ws + 33 * MB);   // (B*S,E) bf16, 1MB
  unsigned short* Wkb = (unsigned short*)(ws + 34 * MB);   // (1024,128) bf16
  unsigned short* Wqb = (unsigned short*)(ws + 35 * MB);
  unsigned short* Wvb = (unsigned short*)(ws + 36 * MB);
  unsigned short* Wpb = (unsigned short*)(ws + 37 * MB);   // (128,1024) bf16

  dim3 blk(32, 1, 1);  // one wave32 per block; EXEC all-ones for WMMA
  const int XN2 = (NROWS * EE) / 2;   // 262144
  const int WN2 = (NCOLS * EE) / 2;   // 65536
  hipLaunchKernelGGL(mla_cvt, dim3((XN2 + 255) / 256), dim3(256), 0, stream,
                     x, (unsigned*)xbf, XN2);
  hipLaunchKernelGGL(mla_cvt, dim3((WN2 + 255) / 256), dim3(256), 0, stream,
                     Wk, (unsigned*)Wkb, WN2);
  hipLaunchKernelGGL(mla_cvt, dim3((WN2 + 255) / 256), dim3(256), 0, stream,
                     Wq, (unsigned*)Wqb, WN2);
  hipLaunchKernelGGL(mla_cvt, dim3((WN2 + 255) / 256), dim3(256), 0, stream,
                     Wv, (unsigned*)Wvb, WN2);
  hipLaunchKernelGGL(mla_cvt, dim3((WN2 + 255) / 256), dim3(256), 0, stream,
                     Wp, (unsigned*)Wpb, WN2);

  hipLaunchKernelGGL(mla_qkv, dim3(NROWS / 16, NCOLS / 64), blk, 0, stream,
                     xbf, Wkb, bk, Wqb, bq, Wvb, bv, Kbf, Qbf, Vt);
  hipLaunchKernelGGL(mla_colstats, dim3(BB * HH, SS / 16), blk, 0, stream,
                     Kbf, Qbf, mS, dS);
  hipLaunchKernelGGL(mla_attnav, dim3(BB * HH, SS / 16), blk, 0, stream,
                     Kbf, Qbf, Vt, mS, dS, Obf);
  hipLaunchKernelGGL(mla_outproj, dim3(NROWS / 16, EE / 64), blk, 0, stream,
                     Obf, Wpb, bp, out);
}
